// TPM_84645215470240
// MI455X (gfx1250) — matrix operand order repaired and checked by reading, not compiled, not run on hardware
//
#include <hip/hip_runtime.h>

#define NPTS 65536

typedef __attribute__((ext_vector_type(16))) _Float16 v16h;
typedef __attribute__((ext_vector_type(8)))  float    v8f;

union FragU { v16h v; unsigned int d[8]; };

__device__ __forceinline__ unsigned int pk2(float a, float b){
  union { _Float16 h[2]; unsigned int u; } x;
  x.h[0] = (_Float16)a; x.h[1] = (_Float16)b;
  return x.u;
}

// A-operand fragment: 16x32 f16 tile stored row-major (packed f16 pairs, strideU uints/row).
// Per ISA: lane m=l&15, hf=l>>4; VGPR j holds K = kb + (j<4?0:16) + hf*8 + (j&3)*2, K+1.
__device__ __forceinline__ v16h fragA(const unsigned int* base, int strideU, int kb, int m, int hf){
  const unsigned int* p = base + m * strideU + ((kb + hf * 8) >> 1);
  FragU u;
#pragma unroll
  for (int j = 0; j < 8; ++j)
    u.d[j] = p[(j < 4) ? j : (4 + j)];   // uint offsets 0..3, 8..11
  return u.v;
}

// B-operand fragment: weights stored D-major (row n = output col, K packed pairs).
// Per ISA (B striped across lanes): lane n=l&15, hf=l>>4; VGPR j holds K = kb + (j<4?0:16) + hf*8 + (j&3)*2, K+1.
__device__ __forceinline__ v16h fragB(const unsigned int* base, int strideU, int kb, int n, int hf){
  const unsigned int* p = base + n * strideU + ((kb + hf * 8) >> 1);
  FragU u;
#pragma unroll
  for (int j = 0; j < 8; ++j) u.d[j] = p[(j < 4) ? j : (4 + j)];
  return u.v;
}

__device__ __forceinline__ v8f wmma_f16(v16h a, v16h b, v8f c){
  return __builtin_amdgcn_wmma_f32_16x16x32_f16(false, a, false, b, (short)0, c, false, false);
}

// Convert a row-major f32 weight matrix W[K][D] into LDS as f16, D-major with
// K padded: dst[d][ku] = pack(W[2ku][d], W[2ku+1][d]); zero-pad d>=D, k>=K.
__device__ __forceinline__ void fill_wt(unsigned int* dst, const float* __restrict__ W,
                                        int D, int Drows, int K, int KpadU,
                                        int tid, int nthr){
  for (int i = tid; i < Drows * KpadU; i += nthr){
    int d = i / KpadU, ku = i - d * KpadU, k = ku * 2;
    float a = (d < D && k       < K) ? W[k * D + d]       : 0.f;
    float b = (d < D && (k + 1) < K) ? W[(k + 1) * D + d] : 0.f;
    dst[i] = pk2(a, b);
  }
}

// ---------------- Kernel A: 729-tap interpolation conv (memory-bound) ----------------
// One wave per point; interp_feat branch collapses exactly to ln2_b[0] (LN over 1 channel).
__global__ __launch_bounds__(256) void k_context(
    const int* __restrict__ nbr_interp, const float* __restrict__ prev_F,
    const float* __restrict__ W_interp, const float* __restrict__ b_interp,
    const float* __restrict__ ln2_b, float2* __restrict__ ctx2){
  int wid  = blockIdx.x * 8 + (threadIdx.x >> 5);
  int lane = threadIdx.x & 31;
  const int* row = nbr_interp + (size_t)wid * 729;
  float acc = 0.f;
  for (int k = lane; k < 729; k += 32){
    __builtin_prefetch(row + k + 96, 0, 1);     // global_prefetch_b8 on the index stream
    int idx = row[k];
    float f = ((unsigned)idx < (unsigned)NPTS) ? prev_F[idx] : 0.f;
    acc = fmaf(W_interp[k], f, acc);
  }
#pragma unroll
  for (int off = 16; off; off >>= 1) acc += __shfl_xor(acc, off, 32);
  if (lane == 0) ctx2[wid] = make_float2(acc + b_interp[0], ln2_b[0]);
}

// ---------------- Kernel B: h = relu(conv27(ctx2, W_conv0)); o = relu(h @ Wi00) --------
__global__ __launch_bounds__(128) void k_conv0(
    const int* __restrict__ nbr3, const float2* __restrict__ ctx2,
    const float* __restrict__ Wc0, const float* __restrict__ bc0,
    const float* __restrict__ Wi00, const float* __restrict__ bi00,
    float* __restrict__ h_g, float* __restrict__ o_g){
  __shared__ __align__(16) unsigned int wA[32 * 32];        // (54->64) x 32  f16
  __shared__ __align__(16) unsigned int wB[16 * 16];        // (32) x (8->16) f16
  __shared__ __align__(16) unsigned int stage[4][16 * 32];  // per-wave 16x64 f16 A tile
  int tid = threadIdx.x;
  fill_wt(wA, Wc0, 32, 32, 54, 32, tid, 128);
  fill_wt(wB, Wi00, 8, 16, 32, 16, tid, 128);
  __syncthreads();
  int w = tid >> 5, lane = tid & 31, col = lane & 15, hf = lane >> 4;
  int n0 = (blockIdx.x * 4 + w) * 16;
  unsigned int* st = stage[w];
  // gather: A[p][k] with k = nbr*2 + {ctx, interp_feat}; one uint per nbr
  for (int j = lane; j < 16 * 32; j += 32){
    int p = j >> 5, q = j & 31;
    unsigned int v = 0u;
    if (q < 27){
      int idx = nbr3[(n0 + p) * 27 + q];
      if ((unsigned)idx < (unsigned)NPTS){
        float2 t = ctx2[idx];
        v = pk2(t.x, t.y);
      }
    }
    st[j] = v;
  }
  __builtin_amdgcn_wave_barrier();
  v8f c0 = {}, c1 = {};
#pragma unroll
  for (int kb = 0; kb < 64; kb += 32){
    v16h a  = fragA(st, 32, kb, col, hf);
    v16h b0 = fragB(wA, 32, kb, col, hf);
    v16h b1 = fragB(wA, 32, kb, col + 16, hf);
    c0 = wmma_f16(a, b0, c0);
    c1 = wmma_f16(a, b1, c1);
  }
  __builtin_amdgcn_wave_barrier();
  float bb0 = bc0[col], bb1 = bc0[col + 16];
#pragma unroll
  for (int r = 0; r < 8; ++r){
    int m = r + hf * 8;
    float v0 = fmaxf(c0[r] + bb0, 0.f);
    float v1 = fmaxf(c1[r] + bb1, 0.f);
    h_g[(n0 + m) * 32 + col]      = v0;
    h_g[(n0 + m) * 32 + col + 16] = v1;
    // repack h tile into A-layout f16 pairs via lane-pairing (even lane packs col,col+1)
    float v0p = __shfl_xor(v0, 1, 32);
    float v1p = __shfl_xor(v1, 1, 32);
    if ((lane & 1) == 0){
      st[m * 32 + (col >> 1)]     = pk2(v0, v0p);
      st[m * 32 + 8 + (col >> 1)] = pk2(v1, v1p);
    }
  }
  __builtin_amdgcn_wave_barrier();
  v16h a2 = fragA(st, 32, 0, col, hf);
  v16h b2 = fragB(wB, 16, 0, col, hf);
  v8f c2 = {};
  c2 = wmma_f16(a2, b2, c2);
  if (col < 8){
    float bo = bi00[col];
#pragma unroll
    for (int r = 0; r < 8; ++r)
      o_g[(n0 + r + hf * 8) * 8 + col] = fmaxf(c2[r] + bo, 0.f);
  }
}

// ---------------- Kernel C: o0 = relu(relu(conv27(o,Wi01)) @ Wi02); o1a = relu(conv27(h,Wi10))
__global__ __launch_bounds__(128) void k_mid(
    const int* __restrict__ nbr3, const float* __restrict__ h_g,
    const float* __restrict__ o_g,
    const float* __restrict__ Wi01, const float* __restrict__ bi01,
    const float* __restrict__ Wi02, const float* __restrict__ bi02,
    const float* __restrict__ Wi10, const float* __restrict__ bi10,
    float* __restrict__ o0_g, float* __restrict__ o1a_g){
  __shared__ __align__(16) unsigned int w01[16 * 112];      // (216->224) x (8->16)
  __shared__ __align__(16) unsigned int w02[16 * 16];       // (8->32) x 16
  __shared__ __align__(16) unsigned int w10[16 * 432];      // (864) x (8->16)
  __shared__ __align__(16) unsigned int stage[4][16 * 16];  // per-wave 16x32 f16 chunk
  int tid = threadIdx.x;
  fill_wt(w01, Wi01, 8, 16, 216, 112, tid, 128);
  fill_wt(w02, Wi02, 16, 16, 8, 16, tid, 128);
  fill_wt(w10, Wi10, 8, 16, 864, 432, tid, 128);
  __syncthreads();
  int w = tid >> 5, lane = tid & 31, col = lane & 15, hf = lane >> 4;
  int n0 = (blockIdx.x * 4 + w) * 16;
  unsigned int* st = stage[w];

  // ---- o0a = relu(conv(o, Wi01)): 7 K-chunks of 4 nbrs x 8 ch ----
  v8f acc = {};
  for (int cb = 0; cb < 7; ++cb){
    int p = lane >> 1;
#pragma unroll
    for (int t = 0; t < 2; ++t){
      int jl = (lane & 1) * 2 + t;
      int nb = cb * 4 + jl;
      float f[8] = {0.f,0.f,0.f,0.f,0.f,0.f,0.f,0.f};
      if (nb < 27){
        int idx = nbr3[(n0 + p) * 27 + nb];
        if ((unsigned)idx < (unsigned)NPTS){
          const float* src = o_g + idx * 8;
#pragma unroll
          for (int c = 0; c < 8; ++c) f[c] = src[c];
        }
      }
      int base = p * 16 + jl * 4;
#pragma unroll
      for (int c = 0; c < 4; ++c) st[base + c] = pk2(f[2 * c], f[2 * c + 1]);
    }
    __builtin_amdgcn_wave_barrier();
    v16h a = fragA(st, 16, 0, col, hf);
    v16h b = fragB(w01, 112, cb * 32, col, hf);
    acc = wmma_f16(a, b, acc);
    __builtin_amdgcn_wave_barrier();
  }
  // ---- o0 = relu(o0a @ Wi02): repack o0a (K=8, zero-pad to 32) then one WMMA ----
  for (int j = lane; j < 16 * 16; j += 32) st[j] = 0u;
  __builtin_amdgcn_wave_barrier();
  float b01 = (col < 8) ? bi01[col] : 0.f;
#pragma unroll
  for (int r = 0; r < 8; ++r){
    int m = r + hf * 8;
    float v = fmaxf(acc[r] + b01, 0.f);
    float vp = __shfl_xor(v, 1, 32);
    if ((lane & 1) == 0 && col < 8)
      st[m * 16 + (col >> 1)] = pk2(v, vp);
  }
  __builtin_amdgcn_wave_barrier();
  {
    v16h a = fragA(st, 16, 0, col, hf);
    v16h b = fragB(w02, 16, 0, col, hf);
    v8f c2 = {};
    c2 = wmma_f16(a, b, c2);
    float b02 = bi02[col];
#pragma unroll
    for (int r = 0; r < 8; ++r)
      o0_g[(n0 + r + hf * 8) * 16 + col] = fmaxf(c2[r] + b02, 0.f);
  }
  __builtin_amdgcn_wave_barrier();

  // ---- o1a = relu(conv(h, Wi10)): 27 K-chunks of 1 nbr x 32 ch ----
  v8f acc1 = {};
  for (int nb = 0; nb < 27; ++nb){
    int p = lane >> 1, cbase = (lane & 1) * 16;
    int idx = nbr3[(n0 + p) * 27 + nb];
    float f[16];
    if ((unsigned)idx < (unsigned)NPTS){
      const float* src = h_g + idx * 32 + cbase;
#pragma unroll
      for (int c = 0; c < 16; ++c) f[c] = src[c];
    } else {
#pragma unroll
      for (int c = 0; c < 16; ++c) f[c] = 0.f;
    }
    int base = p * 16 + (cbase >> 1);
#pragma unroll
    for (int c = 0; c < 8; ++c) st[base + c] = pk2(f[2 * c], f[2 * c + 1]);
    __builtin_amdgcn_wave_barrier();
    v16h a = fragA(st, 16, 0, col, hf);
    v16h b = fragB(w10, 432, nb * 32, col, hf);
    acc1 = wmma_f16(a, b, acc1);
    __builtin_amdgcn_wave_barrier();
  }
  if (col < 8){
    float b10 = bi10[col];
#pragma unroll
    for (int r = 0; r < 8; ++r)
      o1a_g[(n0 + r + hf * 8) * 8 + col] = fmaxf(acc1[r] + b10, 0.f);
  }
}

// ---------------- Kernel D: o1 = relu(conv27(o1a,Wi11)); out = relu(cat(o0,o1)+h) @ W_conv1 + b
__global__ __launch_bounds__(128) void k_final(
    const int* __restrict__ nbr3, const float* __restrict__ h_g,
    const float* __restrict__ o0_g, const float* __restrict__ o1a_g,
    const float* __restrict__ Wi11, const float* __restrict__ bi11,
    const float* __restrict__ Wc1, const float* __restrict__ bc1,
    float* __restrict__ out){
  __shared__ __align__(16) unsigned int w11[16 * 112];      // (216->224) x 16
  __shared__ __align__(16) unsigned int wc1[32 * 16];       // (32) x 32
  __shared__ __align__(16) unsigned int stage[4][16 * 16];
  int tid = threadIdx.x;
  fill_wt(w11, Wi11, 16, 16, 216, 112, tid, 128);
  fill_wt(wc1, Wc1, 32, 32, 32, 16, tid, 128);
  __syncthreads();
  int w = tid >> 5, lane = tid & 31, col = lane & 15, hf = lane >> 4;
  int n0 = (blockIdx.x * 4 + w) * 16;
  unsigned int* st = stage[w];

  v8f acc = {};
  for (int cb = 0; cb < 7; ++cb){
    int p = lane >> 1;
#pragma unroll
    for (int t = 0; t < 2; ++t){
      int jl = (lane & 1) * 2 + t;
      int nb = cb * 4 + jl;
      float f[8] = {0.f,0.f,0.f,0.f,0.f,0.f,0.f,0.f};
      if (nb < 27){
        int idx = nbr3[(n0 + p) * 27 + nb];
        if ((unsigned)idx < (unsigned)NPTS){
          const float* src = o1a_g + idx * 8;
#pragma unroll
          for (int c = 0; c < 8; ++c) f[c] = src[c];
        }
      }
      int base = p * 16 + jl * 4;
#pragma unroll
      for (int c = 0; c < 4; ++c) st[base + c] = pk2(f[2 * c], f[2 * c + 1]);
    }
    __builtin_amdgcn_wave_barrier();
    v16h a = fragA(st, 16, 0, col, hf);
    v16h b = fragB(w11, 112, cb * 32, col, hf);
    acc = wmma_f16(a, b, acc);
    __builtin_amdgcn_wave_barrier();
  }
  // blk = cat(o0, o1) + h ; repack relu(blk) (16x32 f16) via lane pairing
  float b11 = bi11[col];
#pragma unroll
  for (int r = 0; r < 8; ++r){
    int m = r + hf * 8;
    float o1v = fmaxf(acc[r] + b11, 0.f);
    float r0 = fmaxf(o0_g[(n0 + m) * 16 + col] + h_g[(n0 + m) * 32 + col], 0.f);
    float r1 = fmaxf(o1v + h_g[(n0 + m) * 32 + col + 16], 0.f);
    float r0p = __shfl_xor(r0, 1, 32);
    float r1p = __shfl_xor(r1, 1, 32);
    if ((lane & 1) == 0){
      st[m * 16 + (col >> 1)]     = pk2(r0, r0p);
      st[m * 16 + 8 + (col >> 1)] = pk2(r1, r1p);
    }
  }
  __builtin_amdgcn_wave_barrier();
  v16h a2 = fragA(st, 16, 0, col, hf);
  v16h b0 = fragB(wc1, 16, 0, col, hf);
  v16h b1 = fragB(wc1, 16, 0, col + 16, hf);
  v8f d0 = {}, d1 = {};
  d0 = wmma_f16(a2, b0, d0);
  d1 = wmma_f16(a2, b1, d1);
  float bo0 = bc1[col], bo1 = bc1[col + 16];
#pragma unroll
  for (int r = 0; r < 8; ++r){
    int m = r + hf * 8;
    out[(n0 + m) * 32 + col]      = d0[r] + bo0;
    out[(n0 + m) * 32 + col + 16] = d1[r] + bo1;
  }
}

extern "C" void kernel_launch(void* const* d_in, const int* in_sizes, int n_in,
                              void* d_out, int out_size, void* d_ws, size_t ws_size,
                              hipStream_t stream){
  const float* prev_F     = (const float*)d_in[0];
  const int*   nbr_interp = (const int*)  d_in[3];
  const int*   nbr3       = (const int*)  d_in[4];
  const float* W_interp   = (const float*)d_in[6];
  const float* b_interp   = (const float*)d_in[7];
  const float* W_conv0    = (const float*)d_in[8];
  const float* b_conv0    = (const float*)d_in[9];
  const float* W_conv1    = (const float*)d_in[10];
  const float* b_conv1    = (const float*)d_in[11];
  const float* Wi00       = (const float*)d_in[12];
  const float* bi00       = (const float*)d_in[13];
  const float* Wi01       = (const float*)d_in[14];
  const float* bi01       = (const float*)d_in[15];
  const float* Wi02       = (const float*)d_in[16];
  const float* bi02       = (const float*)d_in[17];
  const float* Wi10       = (const float*)d_in[18];
  const float* bi10       = (const float*)d_in[19];
  const float* Wi11       = (const float*)d_in[20];
  const float* bi11       = (const float*)d_in[21];
  const float* ln2_b      = (const float*)d_in[33];
  (void)in_sizes; (void)n_in; (void)out_size; (void)ws_size;

  char* ws = (char*)d_ws;
  float2* ctx2  = (float2*)(ws);                                  // 512 KB
  float*  h_g   = (float*) (ws + (size_t)(1u << 19));             // 8 MB
  float*  o_g   = (float*) (ws + (size_t)(1u << 19) + (8u << 20));  // 2 MB
  float*  o0_g  = (float*) (ws + (size_t)(1u << 19) + (10u << 20)); // 4 MB
  float*  o1a_g = (float*) (ws + (size_t)(1u << 19) + (14u << 20)); // 2 MB

  k_context<<<NPTS / 8, 256, 0, stream>>>(nbr_interp, prev_F, W_interp, b_interp, ln2_b, ctx2);
  k_conv0  <<<NPTS / 64, 128, 0, stream>>>(nbr3, ctx2, W_conv0, b_conv0, Wi00, bi00, h_g, o_g);
  k_mid    <<<NPTS / 64, 128, 0, stream>>>(nbr3, h_g, o_g, Wi01, bi01, Wi02, bi02, Wi10, bi10, o0_g, o1a_g);
  k_final  <<<NPTS / 64, 128, 0, stream>>>(nbr3, h_g, o0_g, o1a_g, Wi11, bi11, W_conv1, b_conv1, (float*)d_out);
}
